// DeformableTransformerDecoderLayer_13606456394035
// MI455X (gfx1250) — matrix operand order repaired and checked by reading, not compiled, not run on hardware
//
#include <hip/hip_runtime.h>
#include <hip/hip_bf16.h>

// ---------------------------------------------------------------------------
// DeformableTransformerDecoderLayer for MI455X (gfx1250, wave32, WMMA)
// bf16 data path end-to-end for GEMMs, f32 accumulation in WMMA.
// Fragment-major LDS layouts -> ds_load_b128 straight into WMMA operands.
// Async global->LDS staging (gfx1250 ASYNCcnt path) when available.
// ---------------------------------------------------------------------------

constexpr int kD    = 256;
constexpr int kDFFN = 1024;
constexpr int kH    = 8;
constexpr int kP    = 4;
constexpr int kL    = 4;
constexpr int kB    = 32;
constexpr int kLQ   = 300;
constexpr int kS    = 3840;
constexpr int kM    = kB * kLQ;       // 9600 query rows
constexpr int kMV   = kB * kS;        // 122880 value rows
constexpr int kLP   = kH * kL * kP;   // 128

__constant__ const int c_lens[4]   = {2048, 1024, 512, 256};
__constant__ const int c_starts[4] = {0, 2048, 3072, 3584};

typedef __attribute__((__ext_vector_type__(16))) __bf16 v16bf;
typedef __attribute__((__ext_vector_type__(8)))  float  v8f;
typedef __attribute__((__ext_vector_type__(4)))  int    v4i;

union alignas(32) Frag16 { unsigned short u[16]; uint4 q[2]; v16bf v; };

#define AS1 __attribute__((address_space(1)))
#define AS3 __attribute__((address_space(3)))

#if defined(__has_builtin)
#  if __has_builtin(__builtin_amdgcn_global_load_async_to_lds_b128) && \
      __has_builtin(__builtin_amdgcn_s_wait_asynccnt)
#    define USE_ASYNC_LDS 1
#  endif
#endif
#ifndef USE_ASYNC_LDS
#  define USE_ASYNC_LDS 0
#endif

__device__ __forceinline__ unsigned short f32_to_bf16(float f) {
    union { float f; unsigned int u; } cv; cv.f = f;
    unsigned int u = cv.u;
    unsigned int r = (u + 0x7FFFu + ((u >> 16) & 1u)) >> 16;  // RNE
    return (unsigned short)r;
}
__device__ __forceinline__ float bf16_to_f32(unsigned short u) {
    union { unsigned int i; float f; } c; c.i = ((unsigned int)u) << 16;
    return c.f;
}

#if USE_ASYNC_LDS
__device__ __forceinline__ void async_copy16(const unsigned short* gsrc,
                                             unsigned short* ldst)
{
    __builtin_amdgcn_global_load_async_to_lds_b128(
        (AS1 v4i*)(unsigned long long)gsrc, (AS3 v4i*)ldst, 0, 0);
}
#endif

// ---------------------------------------------------------------------------
// GEMM: C[M,N] = act(A[M,K](bf16) @ W[K,N](prepped bf16) + bias)
// Block tile 128x128, 8 waves (4x2), wave tile 32x64 = 2x4 WMMA 16x16x32.
// A is plain row-major bf16; W is fragment-major: [K/32][N/16][lane32][e16].
// M%128==0, N%128==0, K%32==0 at all call sites.
// ---------------------------------------------------------------------------
constexpr int BM = 128, BN = 128, BK = 32;

__global__ __launch_bounds__(256) void gemm_bf16_wmma(
    const unsigned short* __restrict__ A,
    const unsigned short* __restrict__ Wp,
    const float* __restrict__ bias,
    float* __restrict__ Cf, unsigned short* __restrict__ Cb,
    int M, int N, int K, int act)
{
    __shared__ alignas(16) unsigned short As[8 * 32 * 16];  // [mtile][lane][16]
    __shared__ alignas(16) unsigned short Bs[8 * 32 * 16];  // [ntile][lane][16]
    (void)M;

    const int tid  = threadIdx.x;
    const int lane = tid & 31;
    const int wave = tid >> 5;
    const int wm   = wave >> 1;   // 0..3
    const int wn   = wave & 1;    // 0..1
    const long block_m = (long)blockIdx.y * BM;
    const long block_n = (long)blockIdx.x * BN;
    const int  ntilesN = N >> 4;
    const int  nt0     = (int)(block_n >> 4);

    v8f acc[2][4];
#pragma unroll
    for (int i = 0; i < 2; ++i)
#pragma unroll
        for (int j = 0; j < 4; ++j)
#pragma unroll
            for (int r = 0; r < 8; ++r) acc[i][j][r] = 0.0f;

    // A staging: thread (wave w, lane l) owns fragment chunk (mtile=w, lane=l):
    //   global row = block_m + w*16 + (l&15); k offsets kb + {0,8}, kb = (l&16)?16:0
    const unsigned short* aRow =
        A + (block_m + wave * 16 + (lane & 15)) * (long)K + ((lane & 16) ? 16 : 0);
    unsigned short* dstA = &As[(wave * 32 + lane) * 16];
    unsigned short* dstB = &Bs[tid * 16];

    for (int k0 = 0; k0 < K; k0 += BK) {
        const unsigned short* srcB =
            Wp + ((((long)(k0 >> 5)) * ntilesN + nt0) << 9) + tid * 16;
#if USE_ASYNC_LDS
        async_copy16(aRow + k0,      dstA);
        async_copy16(aRow + k0 + 8,  dstA + 8);
        async_copy16(srcB,           dstB);
        async_copy16(srcB + 8,       dstB + 8);
        __builtin_amdgcn_s_wait_asynccnt(0);
        __syncthreads();
#else
        *(uint4*)dstA       = *(const uint4*)(aRow + k0);
        *(uint4*)(dstA + 8) = *(const uint4*)(aRow + k0 + 8);
        *(uint4*)dstB       = *(const uint4*)srcB;
        *(uint4*)(dstB + 8) = *(const uint4*)(srcB + 8);
        __syncthreads();
#endif

        Frag16 af[2];
#pragma unroll
        for (int im = 0; im < 2; ++im) {
            const unsigned short* p = &As[((wm * 2 + im) * 32 + lane) * 16];
            af[im].q[0] = *(const uint4*)p;
            af[im].q[1] = *(const uint4*)(p + 8);
        }
        Frag16 bfr[4];
#pragma unroll
        for (int in = 0; in < 4; ++in) {
            const unsigned short* p = &Bs[((wn * 4 + in) * 32 + lane) * 16];
            bfr[in].q[0] = *(const uint4*)p;
            bfr[in].q[1] = *(const uint4*)(p + 8);
        }
#pragma unroll
        for (int im = 0; im < 2; ++im)
#pragma unroll
            for (int in = 0; in < 4; ++in)
                acc[im][in] = __builtin_amdgcn_wmma_f32_16x16x32_bf16(
                    false, af[im].v, false, bfr[in].v,
                    (short)0, acc[im][in], false, false);
        __syncthreads();
    }

    // C/D layout: 8 VGPRs; lane<16 -> M=r, lane>=16 -> M=8+r; N=lane&15
    const int mhalf = (lane >> 4) * 8;
    const int nfr   = lane & 15;
#pragma unroll
    for (int im = 0; im < 2; ++im)
#pragma unroll
        for (int in = 0; in < 4; ++in) {
            const long n = block_n + wn * 64 + in * 16 + nfr;
            const float bv = bias ? bias[n] : 0.0f;
#pragma unroll
            for (int r = 0; r < 8; ++r) {
                const long m = block_m + wm * 32 + im * 16 + mhalf + r;
                float v = acc[im][in][r] + bv;
                if (act) v = fmaxf(v, 0.0f);
                if (Cb) Cb[m * (long)N + n] = f32_to_bf16(v);
                else    Cf[m * (long)N + n] = v;
            }
        }
}

// ---------------------------------------------------------------------------
// Weight prep: fp32 [K,N] -> bf16 fragment-major [K/32][N/16][lane32][e16]
//   lane = (n&15) + (koff&16) ; e = koff&15
// ---------------------------------------------------------------------------
__global__ void weight_prep_kernel(const float* __restrict__ W,
                                   unsigned short* __restrict__ Wp,
                                   int K, int N)
{
    const long i = (long)blockIdx.x * 256 + threadIdx.x;
    if (i >= (long)K * N) return;
    const int k = (int)(i / N), n = (int)(i % N);
    const int kb = k >> 5, koff = k & 31;
    const int nt = n >> 4;
    const int l  = (n & 15) + (koff & 16);
    const int e  = koff & 15;
    Wp[(((long)kb * (N >> 4) + nt) * 32 + l) * 16 + e] = f32_to_bf16(W[i]);
}

// ---------------------------------------------------------------------------
// Elementwise: bf16(a), bf16(a+b)
// ---------------------------------------------------------------------------
__global__ void cvt_bf16_kernel(const float* __restrict__ a,
                                unsigned short* __restrict__ c, long n)
{
    long i = (long)blockIdx.x * 256 + threadIdx.x;
    if (i < n) c[i] = f32_to_bf16(a[i]);
}
__global__ void add_bf16_kernel(const float* __restrict__ a,
                                const float* __restrict__ b,
                                unsigned short* __restrict__ c, long n)
{
    long i = (long)blockIdx.x * 256 + threadIdx.x;
    if (i < n) c[i] = f32_to_bf16(a[i] + b[i]);
}

// ---------------------------------------------------------------------------
// Residual + LayerNorm: outf = LN(resid+t2)*g+b ; optional bf16 copy
// ---------------------------------------------------------------------------
__global__ __launch_bounds__(256) void ln_kernel(
    const float* __restrict__ resid, const float* __restrict__ t2,
    const float* __restrict__ g, const float* __restrict__ bta,
    float* __restrict__ outf, unsigned short* __restrict__ outbf)
{
    __shared__ float red[256];
    const int row = blockIdx.x;
    const int t   = threadIdx.x;
    const float x = resid[(size_t)row * kD + t] + t2[(size_t)row * kD + t];

    red[t] = x;
    __syncthreads();
    for (int s = 128; s > 0; s >>= 1) {
        if (t < s) red[t] += red[t + s];
        __syncthreads();
    }
    const float mean = red[0] * (1.0f / kD);
    __syncthreads();

    const float dx = x - mean;
    red[t] = dx * dx;
    __syncthreads();
    for (int s = 128; s > 0; s >>= 1) {
        if (t < s) red[t] += red[t + s];
        __syncthreads();
    }
    const float var  = red[0] * (1.0f / kD);
    const float rstd = rsqrtf(var + 1e-5f);
    const float y    = dx * rstd * g[t] + bta[t];
    outf[(size_t)row * kD + t] = y;
    if (outbf) outbf[(size_t)row * kD + t] = f32_to_bf16(y);
}

// ---------------------------------------------------------------------------
// MHA with online softmax; fp32 Q/K/V in, bf16 out (feeds out-proj GEMM).
// ---------------------------------------------------------------------------
__global__ __launch_bounds__(128) void mha_kernel(
    const float* __restrict__ qh, const float* __restrict__ kh,
    const float* __restrict__ vh, unsigned short* __restrict__ out)
{
    const int bh = blockIdx.x;
    const int b  = bh >> 3;
    const int h  = bh & 7;
    const float scale = 0.17677669529663687f;  // 1/sqrt(32)

    for (int q = threadIdx.x; q < kLQ; q += blockDim.x) {
        float qv[32];
        const float* qp = qh + ((size_t)(b * kLQ + q)) * kD + h * 32;
#pragma unroll
        for (int d = 0; d < 32; ++d) qv[d] = qp[d] * scale;

        float m = -1e30f, s = 0.0f;
        float o[32];
#pragma unroll
        for (int d = 0; d < 32; ++d) o[d] = 0.0f;

        for (int k = 0; k < kLQ; ++k) {
            const float* kp = kh + ((size_t)(b * kLQ + k)) * kD + h * 32;
            float dot = 0.0f;
#pragma unroll
            for (int d = 0; d < 32; ++d) dot += qv[d] * kp[d];
            const float mn   = fmaxf(m, dot);
            const float corr = __expf(m - mn);
            const float p    = __expf(dot - mn);
            s = s * corr + p;
            const float* vp = vh + ((size_t)(b * kLQ + k)) * kD + h * 32;
#pragma unroll
            for (int d = 0; d < 32; ++d) o[d] = o[d] * corr + p * vp[d];
            m = mn;
        }
        const float inv = 1.0f / s;
        unsigned short* op = out + ((size_t)(b * kLQ + q)) * kD + h * 32;
#pragma unroll
        for (int d = 0; d < 32; ++d) op[d] = f32_to_bf16(o[d] * inv);
    }
}

// ---------------------------------------------------------------------------
// Deformable temporal sampling; value bf16 in, bf16 out.
// ---------------------------------------------------------------------------
__global__ __launch_bounds__(256) void deform_sample_kernel(
    const float* __restrict__ off, const float* __restrict__ attnlog,
    const float* __restrict__ ref, const unsigned short* __restrict__ value,
    unsigned short* __restrict__ out, int boundary)
{
    const int row = blockIdx.x;  // b*LQ + lq
    const int t   = threadIdx.x;
    const int h   = t >> 5;
    const int d   = t & 31;
    const int b   = row / kLQ;

    const float* al = attnlog + (size_t)row * kLP + h * (kL * kP);
    float w16[16];
    float mx = -1e30f;
#pragma unroll
    for (int i = 0; i < 16; ++i) { w16[i] = al[i]; mx = fmaxf(mx, w16[i]); }
    float ssum = 0.0f;
#pragma unroll
    for (int i = 0; i < 16; ++i) { w16[i] = __expf(w16[i] - mx); ssum += w16[i]; }
    const float inv = 1.0f / ssum;

    const float* offp = off + (size_t)row * kLP + h * (kL * kP);
    float acc = 0.0f;

#pragma unroll
    for (int l = 0; l < kL; ++l) {
        const float center = ref[((size_t)row * kL + l) * 2 + 0];
        const float width  = ref[((size_t)row * kL + l) * 2 + 1];
        const int T  = c_lens[l];
        const int s0 = c_starts[l];
        const unsigned short* vbase =
            value + ((size_t)b * kS + s0) * kD + h * 32 + d;
#pragma unroll
        for (int p = 0; p < kP; ++p) {
            const float ov = offp[l * kP + p];
            const float anchor =
                boundary ? (center + ((p < kP / 2) ? -0.5f : 0.5f) * width)
                         : center;
            const float loc = anchor + ov * width * (0.5f / kP);
            const float x  = loc * (float)T - 0.5f;
            const float x0 = floorf(x);
            const float wf = x - x0;
            const int i0 = (int)x0;
            const int i1 = i0 + 1;
            const float m0 = (i0 >= 0 && i0 < T) ? 1.0f : 0.0f;
            const float m1 = (i1 >= 0 && i1 < T) ? 1.0f : 0.0f;
            const int i0c = min(max(i0, 0), T - 1);
            const int i1c = min(max(i1, 0), T - 1);
            const float v0 = bf16_to_f32(vbase[(size_t)i0c * kD]);
            const float v1 = bf16_to_f32(vbase[(size_t)i1c * kD]);
            acc += (w16[l * kP + p] * inv) *
                   (v0 * ((1.0f - wf) * m0) + v1 * (wf * m1));
        }
    }
    out[(size_t)row * kD + h * 32 + d] = f32_to_bf16(acc);
}

// ---------------------------------------------------------------------------
// Host-side orchestration
// ---------------------------------------------------------------------------
static inline void launch_gemm(const unsigned short* A, const unsigned short* Wp,
                               const float* bias, float* Cf, unsigned short* Cb,
                               int M, int N, int K, int act, hipStream_t stream)
{
    dim3 grid(N / BN, M / BM);
    gemm_bf16_wmma<<<grid, 256, 0, stream>>>(A, Wp, bias, Cf, Cb, M, N, K, act);
}

extern "C" void kernel_launch(void* const* d_in, const int* in_sizes, int n_in,
                              void* d_out, int out_size, void* d_ws, size_t ws_size,
                              hipStream_t stream)
{
    (void)in_sizes; (void)n_in; (void)out_size; (void)ws_size;

    const float* tgt  = (const float*)d_in[0];
    const float* qpos = (const float*)d_in[1];
    const float* ref  = (const float*)d_in[2];
    const float* src  = (const float*)d_in[3];
    const float* sa_qw = (const float*)d_in[6];
    const float* sa_kw = (const float*)d_in[7];
    const float* sa_vw = (const float*)d_in[8];
    const float* sa_ow = (const float*)d_in[9];
    const float* ff1_w = (const float*)d_in[10];
    const float* ff2_w = (const float*)d_in[11];
    const float* sa_qb = (const float*)d_in[12];
    const float* sa_kb = (const float*)d_in[13];
    const float* sa_vb = (const float*)d_in[14];
    const float* sa_ob = (const float*)d_in[15];
    const float* ff1_b = (const float*)d_in[16];
    const float* ff2_b = (const float*)d_in[17];
    const float* ca_off_w  = (const float*)d_in[18];
    const float* ca_off_b  = (const float*)d_in[19];
    const float* ca_attn_w = (const float*)d_in[20];
    const float* ca_attn_b = (const float*)d_in[21];
    const float* ca_val_w  = (const float*)d_in[22];
    const float* ca_val_b  = (const float*)d_in[23];
    const float* ca_out_w  = (const float*)d_in[24];
    const float* ca_out_b  = (const float*)d_in[25];
    const float* se_off_w  = (const float*)d_in[26];
    const float* se_off_b  = (const float*)d_in[27];
    const float* se_attn_w = (const float*)d_in[28];
    const float* se_attn_b = (const float*)d_in[29];
    const float* se_val_w  = (const float*)d_in[30];
    const float* se_val_b  = (const float*)d_in[31];
    const float* se_out_w  = (const float*)d_in[32];
    const float* se_out_b  = (const float*)d_in[33];
    const float* ln1_g  = (const float*)d_in[34];
    const float* ln1_b  = (const float*)d_in[35];
    const float* lnse_g = (const float*)d_in[36];
    const float* lnse_b = (const float*)d_in[37];
    const float* ln2_g  = (const float*)d_in[38];
    const float* ln2_b  = (const float*)d_in[39];
    const float* ln3_g  = (const float*)d_in[40];
    const float* ln3_b  = (const float*)d_in[41];

    // -------- workspace carving (256B-aligned) --------
    char* wsb = (char*)d_ws;
    size_t woff = 0;
    auto walloc = [&](size_t bytes) {
        void* p = wsb + woff;
        woff += (bytes + 255) & ~(size_t)255;
        return p;
    };
    float* Qf   = (float*)walloc((size_t)kM * kD * 4);
    float* Kf   = (float*)walloc((size_t)kM * kD * 4);
    float* Vf   = (float*)walloc((size_t)kM * kD * 4);
    float* T2f  = (float*)walloc((size_t)kM * kD * 4);
    float* CURf = (float*)walloc((size_t)kM * kD * 4);
    float* OFFf = (float*)walloc((size_t)kM * kLP * 4);
    float* ATTf = (float*)walloc((size_t)kM * kLP * 4);
    unsigned short* qbf   = (unsigned short*)walloc((size_t)kM * kD * 2);
    unsigned short* tgtbf = (unsigned short*)walloc((size_t)kM * kD * 2);
    unsigned short* mhabf = (unsigned short*)walloc((size_t)kM * kD * 2);  // also sample out
    unsigned short* curbf = (unsigned short*)walloc((size_t)kM * kD * 2);
    unsigned short* srcbf = (unsigned short*)walloc((size_t)kMV * kD * 2);
    unsigned short* valbf = (unsigned short*)walloc((size_t)kMV * kD * 2); // also FFN hidden
    unsigned short* hidbf = valbf;  // kM*kDFFN (9.8M) fits in kMV*kD (31.5M)

    auto wprep = [&](const float* w, int K_, int N_) {
        unsigned short* wp = (unsigned short*)walloc((size_t)K_ * N_ * 2);
        long n = (long)K_ * N_;
        weight_prep_kernel<<<(unsigned)((n + 255) / 256), 256, 0, stream>>>(w, wp, K_, N_);
        return (const unsigned short*)wp;
    };
    const unsigned short* wp_saq  = wprep(sa_qw, kD, kD);
    const unsigned short* wp_sak  = wprep(sa_kw, kD, kD);
    const unsigned short* wp_sav  = wprep(sa_vw, kD, kD);
    const unsigned short* wp_sao  = wprep(sa_ow, kD, kD);
    const unsigned short* wp_ff1  = wprep(ff1_w, kD, kDFFN);
    const unsigned short* wp_ff2  = wprep(ff2_w, kDFFN, kD);
    const unsigned short* wp_coff = wprep(ca_off_w, kD, kLP);
    const unsigned short* wp_catt = wprep(ca_attn_w, kD, kLP);
    const unsigned short* wp_cval = wprep(ca_val_w, kD, kD);
    const unsigned short* wp_cout = wprep(ca_out_w, kD, kD);
    const unsigned short* wp_soff = wprep(se_off_w, kD, kLP);
    const unsigned short* wp_satt = wprep(se_attn_w, kD, kLP);
    const unsigned short* wp_sval = wprep(se_val_w, kD, kD);
    const unsigned short* wp_sout = wprep(se_out_w, kD, kD);

    const long nElem = (long)kM * kD;
    const unsigned eGrid = (unsigned)((nElem + 255) / 256);
    const long nSrc = (long)kMV * kD;
    const unsigned sGrid = (unsigned)((nSrc + 255) / 256);

    cvt_bf16_kernel<<<sGrid, 256, 0, stream>>>(src, srcbf, nSrc);
    cvt_bf16_kernel<<<eGrid, 256, 0, stream>>>(tgt, tgtbf, nElem);

    // ---------------- Self-attention ----------------
    add_bf16_kernel<<<eGrid, 256, 0, stream>>>(tgt, qpos, qbf, nElem);
    launch_gemm(qbf,   wp_saq, sa_qb, Qf, nullptr, kM, kD, kD, 0, stream);
    launch_gemm(qbf,   wp_sak, sa_kb, Kf, nullptr, kM, kD, kD, 0, stream);
    launch_gemm(tgtbf, wp_sav, sa_vb, Vf, nullptr, kM, kD, kD, 0, stream);
    mha_kernel<<<kB * kH, 128, 0, stream>>>(Qf, Kf, Vf, mhabf);
    launch_gemm(mhabf, wp_sao, sa_ob, T2f, nullptr, kM, kD, kD, 0, stream);
    ln_kernel<<<kM, 256, 0, stream>>>(tgt, T2f, ln2_g, ln2_b, CURf, curbf);

    // ---------------- Deformable attention ----------------
    auto deform = [&](const unsigned short* wpo, const float* offb,
                      const unsigned short* wpa, const float* attb,
                      const unsigned short* wpv, const float* valb,
                      const unsigned short* wpu, const float* outb,
                      const float* lng, const float* lnb, int boundary) {
        add_bf16_kernel<<<eGrid, 256, 0, stream>>>(CURf, qpos, qbf, nElem);
        launch_gemm(qbf,   wpo, offb, OFFf, nullptr, kM, kLP, kD, 0, stream);
        launch_gemm(qbf,   wpa, attb, ATTf, nullptr, kM, kLP, kD, 0, stream);
        launch_gemm(srcbf, wpv, valb, nullptr, valbf, kMV, kD, kD, 0, stream);
        deform_sample_kernel<<<kM, 256, 0, stream>>>(OFFf, ATTf, ref, valbf,
                                                     mhabf, boundary);
        launch_gemm(mhabf, wpu, outb, T2f, nullptr, kM, kD, kD, 0, stream);
        ln_kernel<<<kM, 256, 0, stream>>>(CURf, T2f, lng, lnb, CURf, curbf);
    };
    deform(wp_coff, ca_off_b, wp_catt, ca_attn_b, wp_cval, ca_val_b,
           wp_cout, ca_out_b, ln1_g, ln1_b, 0);
    deform(wp_soff, se_off_b, wp_satt, se_attn_b, wp_sval, se_val_b,
           wp_sout, se_out_b, lnse_g, lnse_b, 1);

    // ---------------- FFN ----------------
    launch_gemm(curbf, wp_ff1, ff1_b, nullptr, hidbf, kM, kDFFN, kD, 1, stream);
    launch_gemm(hidbf, wp_ff2, ff2_b, T2f, nullptr, kM, kD, kDFFN, 0, stream);
    ln_kernel<<<kM, 256, 0, stream>>>(CURf, T2f, ln3_g, ln3_b, (float*)d_out, nullptr);
}
